// DeepHamActor_58222576664664
// MI455X (gfx1250) — matrix operand order repaired and checked by reading, not compiled, not run on hardware
//
#include <hip/hip_runtime.h>

#define N_NODES 50000
#define N_EDGES 200000
#define F_INPUT 16
#define DMODEL  512
#define HDIM    256
#define LRALPHA 0.1f
#define CSTR    40   // LDS column stride in bf16 elems (80B: 16B-aligned, bank-conflict free)

typedef __attribute__((ext_vector_type(16))) __bf16 v16bf;
typedef __attribute__((ext_vector_type(8)))  float  v8f;

// ---------------------------------------------------------------------------
// Prep: pad raw vertex features [N,16] fp32 -> [N,32] bf16 (zero padded)
// ---------------------------------------------------------------------------
__global__ void cvt_pad_vertices(const float* __restrict__ v, __bf16* __restrict__ x) {
    int idx = blockIdx.x * blockDim.x + threadIdx.x;
    if (idx >= N_NODES * 32) return;
    int n = idx >> 5, j = idx & 31;
    x[idx] = (j < F_INPUT) ? (__bf16)v[n * F_INPUT + j] : (__bf16)0.0f;
}

// ---------------------------------------------------------------------------
// Prep: W [K, Dc] fp32 row-major -> WT [Dc, Kpad] bf16 (zero padded K)
// ---------------------------------------------------------------------------
__global__ void cvt_transpose_w(const float* __restrict__ W, __bf16* __restrict__ WT,
                                int K, int Dc, int Kpad) {
    int idx = blockIdx.x * blockDim.x + threadIdx.x;
    if (idx >= Dc * Kpad) return;
    int d = idx / Kpad, k = idx % Kpad;
    WT[idx] = (k < K) ? (__bf16)W[(size_t)k * Dc + d] : (__bf16)0.0f;
}

// ---------------------------------------------------------------------------
// WMMA GEMM: C[M,NC] = A[M,K](bf16) * B[K,NC]  (B given transposed: BT[NC,K])
// 256 threads = 8 waves; wave = 16x64 strip (4 x v_wmma_f32_16x16x32_bf16);
// block tile = 32(M) x 256(N).  K % 32 == 0, NC % 256 == 0.
//
// B tiles (256 cols x 32 K bf16) double-buffered in LDS (80B padded column
// stride, bank-conflict free), filled by global_load_async_to_lds_b128 one
// K-step ahead (ASYNCcnt pipelining). A fragments software-pipelined in regs.
// Last K-step peeled so the steady-state loop is branch-free.
// ---------------------------------------------------------------------------
__global__ __launch_bounds__(256)
void gemm_bf16_wmma(const __bf16* __restrict__ A, const __bf16* __restrict__ BT,
                    float* __restrict__ C, int M, int K, int NC) {
    __shared__ __bf16 Bs[2][256 * CSTR];        // 2 x 20 KB

    const int lane = threadIdx.x & 31;
    const int wave = threadIdx.x >> 5;
    const int half = lane >> 4;                 // 0: lanes 0-15, 1: lanes 16-31
    const int l16  = lane & 15;

    const int mBase  = (blockIdx.x * 2 + (wave >> 2)) * 16;
    const int nBlock = blockIdx.y * 256;        // block column base (global)
    const int nBase  = (wave & 3) * 64;         // wave column base (block-local)

    // async-copy mapping: thread handles a 16B chunk (8 bf16) of a column
    const int cchunk = lane & 3;
    const int ccol0  = wave * 32 + (lane >> 2);

    int arow = mBase + l16;
    if (arow >= M) arow = M - 1;                // clamp; EXEC stays all-1 for WMMA
    const __bf16* Aptr = A + (size_t)arow * K;

    auto issueB = [&](int buf, int k0) {
        #pragma unroll
        for (int i = 0; i < 4; ++i) {
            int col = ccol0 + i * 8;
            const __bf16* g = BT + (size_t)(nBlock + col) * K + k0 + cchunk * 8;
            unsigned lds = (unsigned)(uintptr_t)&Bs[buf][col * CSTR + cchunk * 8];
            unsigned long long ga = (unsigned long long)(uintptr_t)g;
            asm volatile("global_load_async_to_lds_b128 %0, %1, off"
                         :: "v"(lds), "v"(ga) : "memory");
        }
    };

    auto loadA = [&](v16bf& a, int k0) {
        ((int4*)&a)[0] = *(const int4*)(Aptr + k0 + half * 8);
        ((int4*)&a)[1] = *(const int4*)(Aptr + k0 + 16 + half * 8);
    };

    v8f acc[4];
    #pragma unroll
    for (int t = 0; t < 4; ++t) acc[t] = (v8f){0.f,0.f,0.f,0.f,0.f,0.f,0.f,0.f};

    auto computeTile = [&](int buf, const v16bf& a) {
        v16bf bfrag[4];
        #pragma unroll
        for (int t = 0; t < 4; ++t) {           // preload all B frags (batched dscnt wait)
            const __bf16* bp = &Bs[buf][(nBase + t * 16 + l16) * CSTR + half * 8];
            ((int4*)&bfrag[t])[0] = *(const int4*)(bp);
            ((int4*)&bfrag[t])[1] = *(const int4*)(bp + 16);
        }
        #pragma unroll
        for (int t = 0; t < 4; ++t)
            acc[t] = __builtin_amdgcn_wmma_f32_16x16x32_bf16(
                false, a, false, bfrag[t], (short)0, acc[t], false, false);
    };

    const int nk = K >> 5;

    // prologue: stage first B tile, prefetch first A fragment
    issueB(0, 0);
    v16bf a_next;
    loadA(a_next, 0);

    // steady state: branch-free, one tile in flight
    for (int kt = 0; kt < nk - 1; ++kt) {
        const int cur = kt & 1;
        issueB(cur ^ 1, (kt + 1) * 32);
        // current tile's 4 async ops retired once only the 4 new ones remain
        asm volatile("s_wait_asynccnt 0x4" ::: "memory");
        __syncthreads();                        // B tile visible to all waves

        v16bf a = a_next;
        loadA(a_next, (kt + 1) * 32);           // software-pipeline next A frag
        computeTile(cur, a);
        __syncthreads();                        // done reading before buffer reuse
    }

    // epilogue: last K-step
    {
        const int cur = (nk - 1) & 1;
        asm volatile("s_wait_asynccnt 0x0" ::: "memory");
        __syncthreads();
        computeTile(cur, a_next);
    }

    // C/D layout: VGPR j -> row mBase + j + half*8, col = nBase + t*16 + l16
    #pragma unroll
    for (int t = 0; t < 4; ++t) {
        int col = nBlock + nBase + t * 16 + l16;
        #pragma unroll
        for (int j = 0; j < 8; ++j) {
            int row = mBase + half * 8 + j;
            if (row < M) C[(size_t)row * NC + col] = acc[t][j];
        }
    }
}

// ---------------------------------------------------------------------------
// Graph norm prep
// ---------------------------------------------------------------------------
__global__ void count_deg(const long long* __restrict__ ei, float* __restrict__ cnt) {
    int e = blockIdx.x * blockDim.x + threadIdx.x;
    if (e >= N_EDGES) return;
    atomicAdd(&cnt[(int)ei[N_EDGES + e]], 1.0f);       // dst row of edge_index
}

__global__ void node_norms(const float* __restrict__ cnt, float* __restrict__ dinv,
                           float* __restrict__ selfn) {
    int n = blockIdx.x * blockDim.x + threadIdx.x;
    if (n >= N_NODES) return;
    float dg = 1.0f + cnt[n];
    dinv[n]  = rsqrtf(dg);
    selfn[n] = 1.0f / dg;
}

__global__ void edge_norms(const long long* __restrict__ ei, const float* __restrict__ dinv,
                           float* __restrict__ en) {
    int e = blockIdx.x * blockDim.x + threadIdx.x;
    if (e >= N_EDGES) return;
    en[e] = dinv[(int)ei[e]] * dinv[(int)ei[N_EDGES + e]];
}

// ---------------------------------------------------------------------------
// Message passing: agg[dst] += h[src] * edge_norm   (block per edge)
// ---------------------------------------------------------------------------
__global__ __launch_bounds__(256)
void scatter_edges(const float* __restrict__ h, const long long* __restrict__ ei,
                   const float* __restrict__ en, float* __restrict__ agg) {
    int e = blockIdx.x;
    int s = (int)ei[e];
    int d = (int)ei[N_EDGES + e];
    float w = en[e];
    const float* hs = h + (size_t)s * DMODEL;
    float*       ad = agg + (size_t)d * DMODEL;
    for (int f = threadIdx.x; f < DMODEL; f += 256)
        atomicAdd(&ad[f], hs[f] * w);
}

// ---------------------------------------------------------------------------
// out = tanh(agg + h*self_norm + b) -> bf16 activations for next layer
// ---------------------------------------------------------------------------
__global__ void finalize_layer(const float* __restrict__ h, const float* __restrict__ agg,
                               const float* __restrict__ selfn, const float* __restrict__ b,
                               __bf16* __restrict__ outbf) {
    int idx = blockIdx.x * blockDim.x + threadIdx.x;
    if (idx >= N_NODES * DMODEL) return;
    int n = idx / DMODEL, f = idx % DMODEL;
    float v = tanhf(agg[idx] + h[idx] * selfn[n] + b[f]);
    outbf[idx] = (__bf16)v;
}

// ---------------------------------------------------------------------------
// Per-node score: leaky_relu(P + bp1) . Wp2 + bp2   (block per node, LDS reduce)
// ---------------------------------------------------------------------------
__global__ __launch_bounds__(256)
void score_nodes(const float* __restrict__ P, const float* __restrict__ bp1,
                 const float* __restrict__ Wp2, const float* __restrict__ bp2,
                 float* __restrict__ sc) {
    __shared__ float s[HDIM];
    int n = blockIdx.x, t = threadIdx.x;
    float v = P[(size_t)n * HDIM + t] + bp1[t];
    v = v > 0.0f ? v : LRALPHA * v;
    s[t] = v * Wp2[t];
    __syncthreads();
    for (int o = HDIM / 2; o > 0; o >>= 1) {
        if (t < o) s[t] += s[t + o];
        __syncthreads();
    }
    if (t == 0) sc[n] = s[0] + bp2[0];
}

// ---------------------------------------------------------------------------
// Neighbor mask of current vertex (+ self)
// ---------------------------------------------------------------------------
__global__ void build_mask(const long long* __restrict__ ei, const int* __restrict__ curp,
                           int* __restrict__ mask) {
    int e = blockIdx.x * blockDim.x + threadIdx.x;
    int cur = curp[0];
    if (e == 0) mask[cur] = 1;
    if (e >= N_EDGES) return;
    if ((int)ei[e] == cur) mask[(int)ei[N_EDGES + e]] = 1;
}

// ---------------------------------------------------------------------------
// Masked softmax over all nodes (single block, 3 passes)
// ---------------------------------------------------------------------------
__global__ __launch_bounds__(1024)
void masked_softmax(const float* __restrict__ sc, const int* __restrict__ mask,
                    float* __restrict__ out) {
    __shared__ float s[1024];
    __shared__ float smax, ssum;
    int t = threadIdx.x;

    float m = -INFINITY;
    for (int i = t; i < N_NODES; i += 1024)
        if (mask[i]) m = fmaxf(m, sc[i]);
    s[t] = m; __syncthreads();
    for (int o = 512; o > 0; o >>= 1) { if (t < o) s[t] = fmaxf(s[t], s[t + o]); __syncthreads(); }
    if (t == 0) smax = s[0];
    __syncthreads();

    float acc = 0.0f;
    for (int i = t; i < N_NODES; i += 1024)
        if (mask[i]) acc += __expf(sc[i] - smax);
    s[t] = acc; __syncthreads();
    for (int o = 512; o > 0; o >>= 1) { if (t < o) s[t] += s[t + o]; __syncthreads(); }
    if (t == 0) ssum = s[0];
    __syncthreads();

    for (int i = t; i < N_NODES; i += 1024)
        out[i] = mask[i] ? __expf(sc[i] - smax) / ssum : 0.0f;
}

// ---------------------------------------------------------------------------
extern "C" void kernel_launch(void* const* d_in, const int* in_sizes, int n_in,
                              void* d_out, int out_size, void* d_ws, size_t ws_size,
                              hipStream_t stream) {
    const float*     vertices = (const float*)d_in[0];
    const long long* ei       = (const long long*)d_in[1];   // [2, E] int64
    const int*       curp     = (const int*)d_in[2];
    const float* W1  = (const float*)d_in[3];  const float* b1  = (const float*)d_in[4];
    const float* W2  = (const float*)d_in[5];  const float* b2  = (const float*)d_in[6];
    const float* W3  = (const float*)d_in[7];  const float* b3  = (const float*)d_in[8];
    const float* Wp1 = (const float*)d_in[9];  const float* bp1 = (const float*)d_in[10];
    const float* Wp2 = (const float*)d_in[11]; const float* bp2 = (const float*)d_in[12];

    char* w = (char*)d_ws;
    size_t o = 0;
    __bf16* bfA = (__bf16*)(w + o); o += (size_t)N_NODES * DMODEL * 2;   // 51.2 MB
    __bf16* bfB = (__bf16*)(w + o); o += (size_t)N_NODES * DMODEL * 2;   // 51.2 MB
    float*  hF  = (float*)(w + o);  o += (size_t)N_NODES * DMODEL * 4;   // 102.4 MB
    float*  agg = (float*)(w + o);  o += (size_t)N_NODES * DMODEL * 4;   // 102.4 MB
    float*  P   = agg;  // predictor activations reuse agg after last scatter
    __bf16* W1T  = (__bf16*)(w + o); o += (size_t)DMODEL * 32 * 2;
    __bf16* W2T  = (__bf16*)(w + o); o += (size_t)DMODEL * DMODEL * 2;
    __bf16* W3T  = (__bf16*)(w + o); o += (size_t)DMODEL * DMODEL * 2;
    __bf16* Wp1T = (__bf16*)(w + o); o += (size_t)HDIM * DMODEL * 2;
    float* cnt    = (float*)(w + o); o += (size_t)N_NODES * 4;
    float* dinv   = (float*)(w + o); o += (size_t)N_NODES * 4;
    float* selfn  = (float*)(w + o); o += (size_t)N_NODES * 4;
    float* en     = (float*)(w + o); o += (size_t)N_EDGES * 4;
    float* scores = (float*)(w + o); o += (size_t)N_NODES * 4;
    int*   mask   = (int*)(w + o);   o += (size_t)N_NODES * 4;

    __bf16* Xbf = bfA;   // [N,32] padded input features live at front of bfA

    // --- weight / input conversion ---
    cvt_pad_vertices<<<(N_NODES * 32 + 255) / 256, 256, 0, stream>>>(vertices, Xbf);
    cvt_transpose_w<<<(DMODEL * 32 + 255) / 256, 256, 0, stream>>>(W1, W1T, F_INPUT, DMODEL, 32);
    cvt_transpose_w<<<(DMODEL * DMODEL + 255) / 256, 256, 0, stream>>>(W2, W2T, DMODEL, DMODEL, DMODEL);
    cvt_transpose_w<<<(DMODEL * DMODEL + 255) / 256, 256, 0, stream>>>(W3, W3T, DMODEL, DMODEL, DMODEL);
    cvt_transpose_w<<<(HDIM * DMODEL + 255) / 256, 256, 0, stream>>>(Wp1, Wp1T, DMODEL, HDIM, DMODEL);

    // --- graph norms ---
    hipMemsetAsync(cnt, 0, (size_t)N_NODES * 4, stream);
    count_deg<<<(N_EDGES + 255) / 256, 256, 0, stream>>>(ei, cnt);
    node_norms<<<(N_NODES + 255) / 256, 256, 0, stream>>>(cnt, dinv, selfn);
    edge_norms<<<(N_EDGES + 255) / 256, 256, 0, stream>>>(ei, dinv, en);

    const dim3 gemmBlk(256);
    const int  mTiles = (N_NODES + 31) / 32;     // 1563
    const dim3 gridD(mTiles, DMODEL / 256);      // NC = 512
    const dim3 gridH(mTiles, HDIM / 256);        // NC = 256

    // --- layer 1 (K = 32 padded) ---
    gemm_bf16_wmma<<<gridD, gemmBlk, 0, stream>>>(Xbf, W1T, hF, N_NODES, 32, DMODEL);
    hipMemsetAsync(agg, 0, (size_t)N_NODES * DMODEL * 4, stream);
    scatter_edges<<<N_EDGES, 256, 0, stream>>>(hF, ei, en, agg);
    finalize_layer<<<(N_NODES * DMODEL + 255) / 256, 256, 0, stream>>>(hF, agg, selfn, b1, bfB);

    // --- layer 2 ---
    gemm_bf16_wmma<<<gridD, gemmBlk, 0, stream>>>(bfB, W2T, hF, N_NODES, DMODEL, DMODEL);
    hipMemsetAsync(agg, 0, (size_t)N_NODES * DMODEL * 4, stream);
    scatter_edges<<<N_EDGES, 256, 0, stream>>>(hF, ei, en, agg);
    finalize_layer<<<(N_NODES * DMODEL + 255) / 256, 256, 0, stream>>>(hF, agg, selfn, b2, bfA);

    // --- layer 3 ---
    gemm_bf16_wmma<<<gridD, gemmBlk, 0, stream>>>(bfA, W3T, hF, N_NODES, DMODEL, DMODEL);
    hipMemsetAsync(agg, 0, (size_t)N_NODES * DMODEL * 4, stream);
    scatter_edges<<<N_EDGES, 256, 0, stream>>>(hF, ei, en, agg);
    finalize_layer<<<(N_NODES * DMODEL + 255) / 256, 256, 0, stream>>>(hF, agg, selfn, b3, bfB);

    // --- predictor ---
    gemm_bf16_wmma<<<gridH, gemmBlk, 0, stream>>>(bfB, Wp1T, P, N_NODES, DMODEL, HDIM);
    score_nodes<<<N_NODES, HDIM, 0, stream>>>(P, bp1, Wp2, bp2, scores);

    hipMemsetAsync(mask, 0, (size_t)N_NODES * 4, stream);
    build_mask<<<(N_EDGES + 255) / 256, 256, 0, stream>>>(ei, curp, mask);
    masked_softmax<<<1, 1024, 0, stream>>>(scores, mask, (float*)d_out);
}
